// PtHeadSelection_11106785428097
// MI455X (gfx1250) — matrix operand order repaired and checked by reading, not compiled, not run on hardware
//
#include <hip/hip_runtime.h>

#define BSZ 4
#define CCH 32
#define LT  256
#define DD  512
#define HH  8
#define RR  64

typedef __bf16 v16bf __attribute__((ext_vector_type(16)));
typedef float  v8f   __attribute__((ext_vector_type(8)));
typedef unsigned int u32x4 __attribute__((ext_vector_type(4)));
typedef int          i32x8 __attribute__((ext_vector_type(8)));
typedef int          i32x4 __attribute__((ext_vector_type(4)));

#if defined(__HIP_DEVICE_COMPILE__) && __has_builtin(__builtin_amdgcn_tensor_load_to_lds)
#define USE_TDM 1
#else
#define USE_TDM 0
#endif

__device__ inline __bf16 f2bf(float x) { return (__bf16)x; }

__device__ inline v8f wmma_bf16(v16bf a, v16bf b, v8f c) {
  // v_wmma_f32_16x16x32_bf16: D = A(16x32) * B(32x16) + C
  return __builtin_amdgcn_wmma_f32_16x16x32_bf16(false, a, false, b, (short)0, c,
                                                 false, false);
}

// ---- fragment loaders (wave32 layouts per CDNA5 ISA 7.12.2) ----
// A fragment from row-major LDS a[m*ld + k], 16x32 bf16 (two b128 per lane).
__device__ inline v16bf frag_A(const __bf16* lds, int ld) {
  int lane = threadIdx.x & 31; int m = lane & 15; int hf = lane >> 4;
  const __bf16* p = lds + m * ld;
  v16bf a;
#pragma unroll
  for (int e = 0; e < 8; ++e) a[e] = p[8 * hf + e];
#pragma unroll
  for (int e = 0; e < 8; ++e) a[8 + e] = p[16 + 8 * hf + e];
  return a;
}
// B fragment where LDS holds B^T: lds[n*ld + k] (two b128 per lane).
__device__ inline v16bf frag_B_T(const __bf16* lds, int ld) {
  int lane = threadIdx.x & 31; int n = lane & 15; int hf = lane >> 4;
  const __bf16* p = lds + n * ld;
  v16bf b;
#pragma unroll
  for (int e = 0; e < 8; ++e) b[e] = p[8 * hf + e];
#pragma unroll
  for (int e = 0; e < 8; ++e) b[8 + e] = p[16 + 8 * hf + e];
  return b;
}

__device__ inline void cvt8_store(__bf16* d, const float* s) {
  float4 x0 = ((const float4*)s)[0];
  float4 x1 = ((const float4*)s)[1];
  d[0] = f2bf(x0.x); d[1] = f2bf(x0.y); d[2] = f2bf(x0.z); d[3] = f2bf(x0.w);
  d[4] = f2bf(x1.x); d[5] = f2bf(x1.y); d[6] = f2bf(x1.z); d[7] = f2bf(x1.w);
}
// transposed store: d[i*ld] = bf16(s[i])  (scatter once at fill time so every
// fragment read is a contiguous b128 pair)
__device__ inline void cvt8_store_t(__bf16* d, const float* s, int ld) {
  float4 x0 = ((const float4*)s)[0];
  float4 x1 = ((const float4*)s)[1];
  d[0 * ld] = f2bf(x0.x); d[1 * ld] = f2bf(x0.y);
  d[2 * ld] = f2bf(x0.z); d[3 * ld] = f2bf(x0.w);
  d[4 * ld] = f2bf(x1.x); d[5 * ld] = f2bf(x1.y);
  d[6 * ld] = f2bf(x1.z); d[7 * ld] = f2bf(x1.w);
}
__device__ inline void cp8_bf(__bf16* d, const __bf16* s) {
  *(uint4*)d = *(const uint4*)s;
}
__device__ inline void cp16_bf_t(__bf16* d, const __bf16* s, int ld) {
  uint4 q0 = ((const uint4*)s)[0], q1 = ((const uint4*)s)[1];
  const __bf16* e0 = (const __bf16*)&q0;
  const __bf16* e1 = (const __bf16*)&q1;
#pragma unroll
  for (int i = 0; i < 8; ++i) d[i * ld] = e0[i];
#pragma unroll
  for (int i = 0; i < 8; ++i) d[(8 + i) * ld] = e1[i];
}

#if USE_TDM
// TDM: DMA a 64-row x 32-col bf16 tile (row stride 512 elem) from a
// 32768x512 bf16 tensor into LDS. D# per CDNA5 ISA 8.3/8.4.
__device__ inline void tdm_load_tile_64x32_bf16(unsigned lds_off, const void* g) {
  unsigned long long ga = (unsigned long long)g;
  u32x4 g0;
  g0[0] = 1u;                                               // count=1 (user D#)
  g0[1] = lds_off;                                          // lds_addr (bytes)
  g0[2] = (unsigned)ga;                                     // global_addr[31:0]
  g0[3] = (unsigned)((ga >> 32) & 0x1FFFFFFu) | (2u << 30); // [56:32] | type=2
  i32x8 g1;
  g1[0] = (int)(1u << 16);        // data_size = 1 -> 2 bytes/elem
  g1[1] = (int)(512u << 16);      // tensor_dim0 = 512 (bits 79:48, low half)
  g1[2] = (int)(32768u << 16);    // tensor_dim1 = 32768 (bits 111:80, low half)
  g1[3] = (int)(32u << 16);       // tile_dim0 = 32 (bits 127:112)
  g1[4] = 64;                     // tile_dim1 = 64, tile_dim2 = 0
  g1[5] = 512;                    // tensor_dim0_stride = 512 (bits 191:160)
  g1[6] = 0;                      // stride hi / tensor_dim1_stride lo
  g1[7] = 0;
  i32x4 z = {0, 0, 0, 0};
#if __clang_major__ >= 23
  i32x8 z8 = {0, 0, 0, 0, 0, 0, 0, 0};
  __builtin_amdgcn_tensor_load_to_lds(g0, g1, z, z, z8, 0);
#else
  __builtin_amdgcn_tensor_load_to_lds(g0, g1, z, z, 0);
#endif
}
#endif

#define V8F_ZERO {0.f,0.f,0.f,0.f,0.f,0.f,0.f,0.f}

// ============================================================================
// K1: qu = rope(qz @ U^T), qv = rope(qz @ V^T), quo = rope_o(qz @ U^T)
//     rows = 32768 ([batch][s] flat), one 64-row x 64-col (= one head) tile/WG.
// ============================================================================
__global__ __launch_bounds__(256)
void proj_rope_kernel(const float* __restrict__ qz, const float* __restrict__ U,
                      const float* __restrict__ V, const float* __restrict__ cosT,
                      const float* __restrict__ sinT,
                      __bf16* __restrict__ qu, __bf16* __restrict__ qv,
                      __bf16* __restrict__ quo, int S) {
  __shared__ __bf16 As[64 * 32], Bu[64 * 32], Bv[64 * 32];
  __shared__ float  Ru[64 * 64], Rv[64 * 64];
  const int t = threadIdx.x, w = t >> 5, lane = t & 31;
  const int tm = w & 3, tn0 = w >> 2, tn1 = tn0 + 2;
  const int m0 = blockIdx.x * 64, h = blockIdx.y;
  const int lrow = t >> 2, lc8 = (t & 3) * 8;
  v8f aU0 = V8F_ZERO, aU1 = V8F_ZERO, aV0 = V8F_ZERO, aV1 = V8F_ZERO;

  for (int k0 = 0; k0 < DD; k0 += 32) {
    cvt8_store(As + lrow * 32 + lc8, qz + (size_t)(m0 + lrow) * DD + k0 + lc8);
    cvt8_store(Bu + lrow * 32 + lc8, U + (size_t)(h * 64 + lrow) * DD + k0 + lc8);
    cvt8_store(Bv + lrow * 32 + lc8, V + (size_t)(h * 64 + lrow) * DD + k0 + lc8);
    if (k0 + 32 < DD)  // global_prefetch_b8
      __builtin_prefetch(qz + (size_t)(m0 + lrow) * DD + k0 + 32 + lc8, 0, 1);
    __syncthreads();
    v16bf a  = frag_A(As + tm * 16 * 32, 32);
    v16bf u0 = frag_B_T(Bu + tn0 * 16 * 32, 32);
    v16bf u1 = frag_B_T(Bu + tn1 * 16 * 32, 32);
    v16bf w0 = frag_B_T(Bv + tn0 * 16 * 32, 32);
    v16bf w1 = frag_B_T(Bv + tn1 * 16 * 32, 32);
    aU0 = wmma_bf16(a, u0, aU0);
    aU1 = wmma_bf16(a, u1, aU1);
    aV0 = wmma_bf16(a, w0, aV0);
    aV1 = wmma_bf16(a, w1, aV1);
    __syncthreads();
  }
  {  // stage accumulators (D layout: n = lane&15, m = r + 8*(lane>>4))
    int n = lane & 15, hf = lane >> 4;
    int c0 = tn0 * 16 + n, c1 = tn1 * 16 + n;
#pragma unroll
    for (int r = 0; r < 8; ++r) {
      int m = tm * 16 + r + 8 * hf;
      Ru[m * 64 + c0] = aU0[r]; Ru[m * 64 + c1] = aU1[r];
      Rv[m * 64 + c0] = aV0[r]; Rv[m * 64 + c1] = aV1[r];
    }
  }
  __syncthreads();
  {  // fused RoPE epilogue + bf16 store, layout [batch][h][s][r]
    int rr = t >> 2, c16 = (t & 3) * 16;
    int gr = m0 + rr, b_ = gr / S, s = gr % S;
    size_t ob = (((size_t)b_ * HH + h) * S + s) * (size_t)RR;
#pragma unroll
    for (int e = 0; e < 16; ++e) {
      int r = c16 + e;
      float cs = cosT[s * RR + r], sn = sinT[s * RR + r];
      float u  = Ru[rr * 64 + r];
      float ur = (r < 32) ? -Ru[rr * 64 + r + 32] : Ru[rr * 64 + r - 32];
      float v  = Rv[rr * 64 + r];
      float vr = (r < 32) ? -Rv[rr * 64 + r + 32] : Rv[rr * 64 + r - 32];
      qu[ob + r]  = f2bf(u * cs + ur * sn);
      quo[ob + r] = f2bf(u * cs - ur * sn);
      qv[ob + r]  = f2bf(v * cs + vr * sn);
    }
  }
}

// ============================================================================
// K2: F = qu @ qv^T + mask, batched over bh = batch*H. 32x32 tile / WG.
//     Written straight into the qh output region (softmaxed in place later).
// ============================================================================
__global__ __launch_bounds__(128)
void f_kernel(const __bf16* __restrict__ qu, const __bf16* __restrict__ qv,
              const float* __restrict__ mask, float* __restrict__ outF, int S) {
  __shared__ __bf16 As[32 * 32], Bs[32 * 32];
  const int t = threadIdx.x, w = t >> 5, lane = t & 31;
  const int tm = w & 1, tn = w >> 1;
  const int m0 = blockIdx.x * 32, n0 = blockIdx.y * 32, bh = blockIdx.z;
  const __bf16* Ab = qu + (size_t)bh * S * RR;
  const __bf16* Bb = qv + (size_t)bh * S * RR;
  const int lrow = t >> 2, lc8 = (t & 3) * 8;
  v8f acc = V8F_ZERO;
  for (int k0 = 0; k0 < RR; k0 += 32) {
    cp8_bf(As + lrow * 32 + lc8, Ab + (size_t)(m0 + lrow) * RR + k0 + lc8);
    cp8_bf(Bs + lrow * 32 + lc8, Bb + (size_t)(n0 + lrow) * RR + k0 + lc8);
    __syncthreads();
    v16bf a = frag_A(As + tm * 16 * 32, 32);
    v16bf b = frag_B_T(Bs + tn * 16 * 32, 32);  // qv rows are B^T directly
    acc = wmma_bf16(a, b, acc);
    __syncthreads();
  }
  const float* mb = mask + (size_t)(bh / HH) * S * S;
  float* ob = outF + (size_t)bh * S * S;
  int n = lane & 15, hf = lane >> 4, col = n0 + tn * 16 + n;
#pragma unroll
  for (int r = 0; r < 8; ++r) {
    int m = m0 + tm * 16 + r + 8 * hf;
    ob[(size_t)m * S + col] = acc[r] + mb[(size_t)m * S + col];
  }
}

// ============================================================================
// K3: in-place softmax over concat row (256 time + 32 channel), scale 512.
//     One wave per row; 262144 rows total.
// ============================================================================
__global__ __launch_bounds__(256)
void softmax_kernel(float* __restrict__ qh_t, float* __restrict__ qh_c) {
  const int t = threadIdx.x, w = t >> 5, lane = t & 31;
  int row = blockIdx.x * 8 + w;            // ((b*H + h)*C + c)*L + l
  int l = row & (LT - 1);
  int c = (row >> 8) & (CCH - 1);
  int h = (row >> 13) & (HH - 1);
  int b = row >> 16;
  float* tp = qh_t + (((size_t)(b * CCH + c) * HH + h) << 16) + ((size_t)l << 8);
  float* cp = qh_c + (((size_t)(b * LT + l) * HH + h) << 10) + ((size_t)c << 5);
  float tv[8];
#pragma unroll
  for (int i = 0; i < 8; ++i) tv[i] = tp[lane + 32 * i];
  float cv = cp[lane];
  float mx = cv;
#pragma unroll
  for (int i = 0; i < 8; ++i) mx = fmaxf(mx, tv[i]);
#pragma unroll
  for (int off = 16; off > 0; off >>= 1) mx = fmaxf(mx, __shfl_xor(mx, off, 32));
  float sum = 0.f;
#pragma unroll
  for (int i = 0; i < 8; ++i) { tv[i] = __expf(512.0f * (tv[i] - mx)); sum += tv[i]; }
  cv = __expf(512.0f * (cv - mx)); sum += cv;
#pragma unroll
  for (int off = 16; off > 0; off >>= 1) sum += __shfl_xor(sum, off, 32);
  float inv = 1.0f / sum;
#pragma unroll
  for (int i = 0; i < 8; ++i) tp[lane + 32 * i] = tv[i] * inv;
  cp[lane] = cv * inv;
}

// ============================================================================
// K4: materialize qh_ch = qh_ch_out.transpose(0,3,1,2,4).reshape(flat)
// ============================================================================
__global__ __launch_bounds__(256)
void perm_qh_ch_kernel(const float* __restrict__ src, float* __restrict__ dst) {
  int f = blockIdx.x * 256 + threadIdx.x;  // dims (b, c, l, h, j)
  int j = f & 31, h = (f >> 5) & 7, l = (f >> 8) & 255, c = (f >> 16) & 31, b = f >> 21;
  dst[f] = src[((((size_t)(b * LT + l) * HH + h) * CCH + c) << 5) + j];
}

// ============================================================================
// K5: v1 = rope_o(qh @ qv), v2 = rope(qh^T @ quo), batched over bh.
//     All LDS tiles stored so fragment reads are contiguous b128 pairs.
// ============================================================================
__global__ __launch_bounds__(128)
void av_rope_kernel(const float* __restrict__ qh, const __bf16* __restrict__ qv,
                    const __bf16* __restrict__ quo, const float* __restrict__ cosT,
                    const float* __restrict__ sinT, __bf16* __restrict__ v1o,
                    __bf16* __restrict__ v2o, int S) {
  __shared__ __bf16 As1[32 * 32], As2[32 * 32];   // [m][k]
  __shared__ __bf16 Bs1[64 * 32], Bs2[64 * 32];   // [n][k] (B^T)
  __shared__ float  R1[32 * 64], R2[32 * 64];
  const int t = threadIdx.x, w = t >> 5, lane = t & 31;
  const int tm = w & 1, tnA = w >> 1;
  const int m0 = blockIdx.x * 32, bh = blockIdx.y;
  const float*  qb = qh  + (size_t)bh * S * S;
  const __bf16* vb = qv  + (size_t)bh * S * RR;
  const __bf16* ub = quo + (size_t)bh * S * RR;
  const int lrow = t >> 2, lc8 = (t & 3) * 8, bc16 = (t & 3) * 16;
  v8f a1[2] = {V8F_ZERO, V8F_ZERO}, a2[2] = {V8F_ZERO, V8F_ZERO};

  for (int k0 = 0; k0 < S; k0 += 32) {
    // As1[m][k] = qh[m0+m][k0+k] (row-major copy)
    cvt8_store(As1 + lrow * 32 + lc8, qb + (size_t)(m0 + lrow) * S + k0 + lc8);
    // As2[m][k] = qh[k0+k][m0+m] (transposed scatter-store)
    cvt8_store_t(As2 + lc8 * 32 + lrow, qb + (size_t)(k0 + lrow) * S + m0 + lc8, 32);
    // Bs[n][k] = qv/quo[k0+k][n]   (transposed scatter-store)
    cp16_bf_t(Bs1 + bc16 * 32 + lrow, vb + (size_t)(k0 + lrow) * RR + bc16, 32);
    cp16_bf_t(Bs2 + bc16 * 32 + lrow, ub + (size_t)(k0 + lrow) * RR + bc16, 32);
    __syncthreads();
    v16bf fa1 = frag_A(As1 + tm * 16 * 32, 32);
    v16bf fa2 = frag_A(As2 + tm * 16 * 32, 32);
#pragma unroll
    for (int j = 0; j < 2; ++j) {
      int tn = tnA + 2 * j;
      v16bf fb1 = frag_B_T(Bs1 + tn * 16 * 32, 32);
      v16bf fb2 = frag_B_T(Bs2 + tn * 16 * 32, 32);
      a1[j] = wmma_bf16(fa1, fb1, a1[j]);
      a2[j] = wmma_bf16(fa2, fb2, a2[j]);
    }
    __syncthreads();
  }
  {
    int n = lane & 15, hf = lane >> 4;
#pragma unroll
    for (int j = 0; j < 2; ++j) {
      int col = (tnA + 2 * j) * 16 + n;
#pragma unroll
      for (int r = 0; r < 8; ++r) {
        int m = tm * 16 + r + 8 * hf;
        R1[m * 64 + col] = a1[j][r];
        R2[m * 64 + col] = a2[j][r];
      }
    }
  }
  __syncthreads();
  {
    int rr = t >> 2, c16 = (t & 3) * 16;
    int s = m0 + rr, b_ = bh / HH, h = bh % HH;
    size_t ob = ((size_t)b_ * S + s) * (size_t)DD + h * 64;
#pragma unroll
    for (int e = 0; e < 16; ++e) {
      int r = c16 + e;
      float cs = cosT[s * RR + r], sn = sinT[s * RR + r];
      float x1 = R1[rr * 64 + r];
      float x1r = (r < 32) ? -R1[rr * 64 + r + 32] : R1[rr * 64 + r - 32];
      float x2 = R2[rr * 64 + r];
      float x2r = (r < 32) ? -R2[rr * 64 + r + 32] : R2[rr * 64 + r - 32];
      v1o[ob + r] = f2bf(x1 * cs - x1r * sn);  // rope_apply_o
      v2o[ob + r] = f2bf(x2 * cs + x2r * sn);  // rope_apply
    }
  }
}

// ============================================================================
// K6: G = v1 @ W1 + v2 @ W2  (32768x512 @ 512x512, f32 out straight to d_out)
//     A tiles via TDM (tensor_load_to_lds + s_wait_tensorcnt), overlapping the
//     DMA with the f32->bf16 conversion of the W tiles.
// ============================================================================
__global__ __launch_bounds__(256)
void out_gemm_kernel(const __bf16* __restrict__ v1, const __bf16* __restrict__ v2,
                     const float* __restrict__ W1, const float* __restrict__ W2,
                     float* __restrict__ out) {
  __shared__ __bf16 A1[64 * 32], A2[64 * 32];   // [m][k]
  __shared__ __bf16 B1[64 * 32], B2[64 * 32];   // [n][k] (B^T)
  const int t = threadIdx.x, w = t >> 5, lane = t & 31;
  const int tm = w & 3, tnA = w >> 2;
  const int m0 = blockIdx.x * 64, n0 = blockIdx.y * 64;
  const int br = t >> 3, bc8 = (t & 7) * 8;
#if USE_TDM
  // low 32 bits of a generic LDS pointer = LDS byte offset
  const unsigned offA1 = (unsigned)(unsigned long long)&A1[0];
  const unsigned offA2 = (unsigned)(unsigned long long)&A2[0];
#else
  const int ar = t >> 2, ac8 = (t & 3) * 8;
#endif
  v8f acc[2] = {V8F_ZERO, V8F_ZERO};
  for (int k0 = 0; k0 < DD; k0 += 32) {
#if USE_TDM
    if (w == 0) {  // one wave issues the tile DMAs (EXEC ignored by TDM)
      tdm_load_tile_64x32_bf16(offA1, v1 + (size_t)m0 * DD + k0);
      tdm_load_tile_64x32_bf16(offA2, v2 + (size_t)m0 * DD + k0);
    }
#else
    cp8_bf(A1 + ar * 32 + ac8, v1 + (size_t)(m0 + ar) * DD + k0 + ac8);
    cp8_bf(A2 + ar * 32 + ac8, v2 + (size_t)(m0 + ar) * DD + k0 + ac8);
#endif
    // B^T[n][k] = W[k0+k][n0+n] (transposed scatter; overlaps with TDM)
    cvt8_store_t(B1 + bc8 * 32 + br, W1 + (size_t)(k0 + br) * DD + n0 + bc8, 32);
    cvt8_store_t(B2 + bc8 * 32 + br, W2 + (size_t)(k0 + br) * DD + n0 + bc8, 32);
    if (k0 + 32 < DD)
      __builtin_prefetch(W1 + (size_t)(k0 + 32 + br) * DD + n0 + bc8, 0, 1);
#if USE_TDM
    if (w == 0) __builtin_amdgcn_s_wait_tensorcnt(0);
#endif
    __syncthreads();
    v16bf fa1 = frag_A(A1 + tm * 16 * 32, 32);
    v16bf fa2 = frag_A(A2 + tm * 16 * 32, 32);
#pragma unroll
    for (int j = 0; j < 2; ++j) {
      int tn = tnA + 2 * j;
      v16bf fb1 = frag_B_T(B1 + tn * 16 * 32, 32);
      v16bf fb2 = frag_B_T(B2 + tn * 16 * 32, 32);
      acc[j] = wmma_bf16(fa1, fb1, acc[j]);
      acc[j] = wmma_bf16(fa2, fb2, acc[j]);
    }
    __syncthreads();
  }
  int n = lane & 15, hf = lane >> 4;
#pragma unroll
  for (int j = 0; j < 2; ++j) {
    int col = n0 + (tnA + 2 * j) * 16 + n;
#pragma unroll
    for (int r = 0; r < 8; ++r) {
      int m = m0 + tm * 16 + r + 8 * hf;
      out[(size_t)m * DD + col] = acc[j][r];
    }
  }
}

// ============================================================================
extern "C" void kernel_launch(void* const* d_in, const int* in_sizes, int n_in,
                              void* d_out, int out_size, void* d_ws, size_t ws_size,
                              hipStream_t stream) {
  const float* qz     = (const float*)d_in[0];
  const float* mask_t = (const float*)d_in[1];
  const float* mask_c = (const float*)d_in[2];
  const float* cos_t  = (const float*)d_in[3];
  const float* sin_t  = (const float*)d_in[4];
  const float* cos_c  = (const float*)d_in[5];
  const float* sin_c  = (const float*)d_in[6];
  const float* u_t    = (const float*)d_in[7];
  const float* v_t    = (const float*)d_in[8];
  const float* u_c    = (const float*)d_in[9];
  const float* v_c    = (const float*)d_in[10];

  float* out  = (float*)d_out;
  float* G_t  = out;                 // (4,32,256,512)
  float* G_c  = out + 16777216;      // (4,32,256,512) (flat-preserving reshape)
  float* qh_t = out + 33554432;      // (4,32,8,256,256) == [bh][i][j]
  float* qh_c = out + 100663296;     // (4,256,8,32,32)  == [bh_c][c][j]

  const size_t BFB = 33554432;       // bytes per 32768x512 bf16 array
  char* wsp = (char*)d_ws;
  __bf16* qu_t  = (__bf16*)(wsp + 0 * BFB);
  __bf16* qv_t  = (__bf16*)(wsp + 1 * BFB);
  __bf16* quo_t = (__bf16*)(wsp + 2 * BFB);
  __bf16* qu_c  = (__bf16*)(wsp + 3 * BFB);
  __bf16* qv_c  = (__bf16*)(wsp + 4 * BFB);
  __bf16* quo_c = (__bf16*)(wsp + 5 * BFB);
  __bf16* v2_t  = (__bf16*)(wsp + 6 * BFB);
  __bf16* v2_c  = (__bf16*)(wsp + 7 * BFB);
  float*  qh_cp = (float*) (wsp + 8 * BFB);  // permuted channel weights (f32)
  __bf16* v1_t  = qu_t;   // alias: qu_t dead after F_t
  __bf16* v1_c  = qu_c;   // alias: qu_c dead after F_c

  // projections + RoPE (time view S=256, flat channel view S=32; same rows)
  proj_rope_kernel<<<dim3(512, 8), 256, 0, stream>>>(qz, u_t, v_t, cos_t, sin_t,
                                                     qu_t, qv_t, quo_t, 256);
  proj_rope_kernel<<<dim3(512, 8), 256, 0, stream>>>(qz, u_c, v_c, cos_c, sin_c,
                                                     qu_c, qv_c, quo_c, 32);
  // logits straight into d_out qh regions
  f_kernel<<<dim3(8, 8, 1024), 128, 0, stream>>>(qu_t, qv_t, mask_t, qh_t, 256);
  f_kernel<<<dim3(1, 1, 8192), 128, 0, stream>>>(qu_c, qv_c, mask_c, qh_c, 32);
  // in-place softmax over 288-wide concatenated rows
  softmax_kernel<<<32768, 256, 0, stream>>>(qh_t, qh_c);
  // reference's flat-reinterpreting permutation of channel weights
  perm_qh_ch_kernel<<<32768, 256, 0, stream>>>(qh_c, qh_cp);
  // v1/v2 with fused RoPE
  av_rope_kernel<<<dim3(8, 1024), 128, 0, stream>>>(qh_t, qv_t, quo_t, cos_t,
                                                    sin_t, v1_t, v2_t, 256);
  av_rope_kernel<<<dim3(1, 8192), 128, 0, stream>>>(qh_cp, qv_c, quo_c, cos_c,
                                                    sin_c, v1_c, v2_c, 32);
  // final dual-GEMM into d_out
  out_gemm_kernel<<<dim3(512, 8), 256, 0, stream>>>(v1_t, v2_t, u_t, v_t, G_t);
  out_gemm_kernel<<<dim3(512, 8), 256, 0, stream>>>(v1_c, v2_c, u_c, v_c, G_c);
}
